// HumanObjectMatcher_3710851744270
// MI455X (gfx1250) — matrix operand order repaired and checked by reading, not compiled, not run on hardware
//
#include <hip/hip_runtime.h>
#include <hip/hip_bf16.h>
#include <math.h>

// ---------------------------------------------------------------------------
// CDNA5 (gfx1250) implementation. Heavy GEMMs use v_wmma_f32_16x16x32_f16
// (wave32, 16x16x32, f32 accum) with double-buffered f16 LDS staging.
// ---------------------------------------------------------------------------

typedef _Float16 v16h   __attribute__((ext_vector_type(16)));
typedef _Float16 half8  __attribute__((ext_vector_type(8)));
typedef _Float16 half4h __attribute__((ext_vector_type(4)));
typedef float    v8f    __attribute__((ext_vector_type(8)));

#define BATCH   16
#define NTOK    128
#define NHKEEP  32
#define NPAIR   4064              // 32*127
#define FROWS   (BATCH * NPAIR)   // 65024
#define DM      256
#define MPOSE   64
#define NHEADS  8
#define DH      32

// p -> (x, y) for keep = (x != y) & (x < 32), row-major flatten order
__device__ __forceinline__ void pair_xy(int p, int& x, int& y) {
    x = p / 127;
    int r = p - x * 127;
    y = r + (r >= x ? 1 : 0);
}

// ---------------------------------------------------------------------------
// WMMA GEMM: C[r, n] = act( sum_k A[r*lda + k] * W[k*ldw + n] + bias[n] )
// Requirements: K % 32 == 0, N % 64 == 0 (rows arbitrary; A row index clamped).
// Block tile 64x64, 128 threads (4 waves), wave -> 32x32 (2x2 wmma tiles).
// Double-buffered LDS staging, fp32 -> f16 conversion on the fly.
// ---------------------------------------------------------------------------
__global__ __launch_bounds__(128) void gemm_wmma(
    const float* __restrict__ A, const float* __restrict__ W,
    const float* __restrict__ bias, float* __restrict__ C,
    int rows, int K, int N, int lda, int ldw, int ldc, int act)
{
    __shared__ __align__(16) _Float16 As[2][64 * 32];  // [row][k]  (row stride 32)
    __shared__ __align__(16) _Float16 Bt[2][64 * 32];  // [n][k]    (transposed)

    const int t    = threadIdx.x;
    const int lane = t & 31;
    const int wv   = t >> 5;
    const int wm   = wv >> 1;
    const int wn   = wv & 1;
    const int row0 = blockIdx.y * 64;
    const int col0 = blockIdx.x * 64;
    const int half = lane >> 4;    // K-chunk selector
    const int mloc = lane & 15;    // m (A) / n (B,C) within 16

    // staging decode (computed once)
    // A: 512 float4s: idx=i*128+t -> ar = idx>>3 (0..63), ac4 = idx&7 (col = ac4*4)
    // B: 512 float4s: idx=i*128+t -> bk = idx>>4 (0..31), bn4 = idx&15 (n = bn4*4)
    int ar[4], ac[4], bk[4], bn[4];
    const float* aptr[4];
    #pragma unroll
    for (int i = 0; i < 4; ++i) {
        int idx = i * 128 + t;
        ar[i] = idx >> 3; ac[i] = (idx & 7) * 4;
        int grow = row0 + ar[i];
        grow = grow < rows ? grow : rows - 1;          // clamp: no branch on load
        aptr[i] = A + (size_t)grow * lda + ac[i];
        bk[i] = idx >> 4; bn[i] = (idx & 15) * 4;
    }

    v8f acc[2][2];
    #pragma unroll
    for (int mt = 0; mt < 2; ++mt)
        #pragma unroll
        for (int nt = 0; nt < 2; ++nt)
            #pragma unroll
            for (int r = 0; r < 8; ++r) acc[mt][nt][r] = 0.f;

    float4 ra[4], rb[4];
    // prologue: load tile 0
    #pragma unroll
    for (int i = 0; i < 4; ++i) {
        ra[i] = *(const float4*)(aptr[i]);
        rb[i] = *(const float4*)(W + (size_t)bk[i] * ldw + col0 + bn[i]);
    }

    int buf = 0;
    for (int k0 = 0; k0 < K; k0 += 32) {
        // store staged regs -> LDS[buf]
        #pragma unroll
        for (int i = 0; i < 4; ++i) {
            half4h ha = { (_Float16)ra[i].x, (_Float16)ra[i].y,
                          (_Float16)ra[i].z, (_Float16)ra[i].w };
            *(half4h*)&As[buf][ar[i] * 32 + ac[i]] = ha;
            Bt[buf][(bn[i] + 0) * 32 + bk[i]] = (_Float16)rb[i].x;
            Bt[buf][(bn[i] + 1) * 32 + bk[i]] = (_Float16)rb[i].y;
            Bt[buf][(bn[i] + 2) * 32 + bk[i]] = (_Float16)rb[i].z;
            Bt[buf][(bn[i] + 3) * 32 + bk[i]] = (_Float16)rb[i].w;
        }
        __syncthreads();

        // issue next tile's global loads (hidden behind WMMA)
        if (k0 + 32 < K) {
            int k1 = k0 + 32;
            #pragma unroll
            for (int i = 0; i < 4; ++i) {
                ra[i] = *(const float4*)(aptr[i] + k1);
                rb[i] = *(const float4*)(W + (size_t)(k1 + bk[i]) * ldw + col0 + bn[i]);
            }
        }

        // fragments (ISA 7.12.2 layouts), two 128-bit LDS loads each
        v16h afr[2], bfr[2];
        #pragma unroll
        for (int mt = 0; mt < 2; ++mt) {
            const half8* p = (const half8*)&As[buf][(wm * 32 + mt * 16 + mloc) * 32];
            half8 lo = p[half];
            half8 hi = p[2 + half];
            #pragma unroll
            for (int e = 0; e < 8; ++e) { afr[mt][e] = lo[e]; afr[mt][e + 8] = hi[e]; }
        }
        #pragma unroll
        for (int nt = 0; nt < 2; ++nt) {
            const half8* p = (const half8*)&Bt[buf][(wn * 32 + nt * 16 + mloc) * 32];
            half8 lo = p[half];
            half8 hi = p[2 + half];
            #pragma unroll
            for (int e = 0; e < 8; ++e) { bfr[nt][e] = lo[e]; bfr[nt][e + 8] = hi[e]; }
        }
        #pragma unroll
        for (int mt = 0; mt < 2; ++mt)
            #pragma unroll
            for (int nt = 0; nt < 2; ++nt)
                acc[mt][nt] = __builtin_amdgcn_wmma_f32_16x16x32_f16(
                    false, afr[mt], false, bfr[nt], (short)0, acc[mt][nt], false, false);
        buf ^= 1;
    }

    // epilogue: C/D layout: lane n=mloc, vgpr r -> m = r + 8*half
    #pragma unroll
    for (int mt = 0; mt < 2; ++mt)
        #pragma unroll
        for (int nt = 0; nt < 2; ++nt) {
            int gcol = col0 + wn * 32 + nt * 16 + mloc;
            float bv = bias[gcol];
            #pragma unroll
            for (int r = 0; r < 8; ++r) {
                int grow = row0 + wm * 32 + mt * 16 + r + 8 * half;
                if (grow < rows) {
                    float v = acc[mt][nt][r] + bv;
                    if (act) v = fmaxf(v, 0.f);
                    C[(size_t)grow * ldc + gcol] = v;
                }
            }
        }
}

// ---------------------------------------------------------------------------
// Fused softmax attention: one wave per (query row, head). dh=32, S in {64,128}.
// ---------------------------------------------------------------------------
__global__ __launch_bounds__(128) void attn_kernel(
    const float* __restrict__ Q, const float* __restrict__ Kb,
    const float* __restrict__ Vb, float* __restrict__ O,
    int rowsTotal, int Tq, int S, float scale)
{
    int task = blockIdx.x * 4 + (threadIdx.x >> 5);
    if (task >= rowsTotal * NHEADS) return;
    int lane = threadIdx.x & 31;
    int row = task >> 3, hd = task & 7;
    int b = row / Tq;
    const float* qp = Q + (size_t)row * DM + hd * DH;
    float qown = qp[lane];
    const float* kb = Kb + (size_t)b * S * DM + hd * DH;
    const float* vb = Vb + (size_t)b * S * DM + hd * DH;
    int nch = S >> 5;
    float sc[4];
    for (int j = 0; j < nch; ++j) {
        const float* kp = kb + (size_t)(j * 32 + lane) * DM;
        float d = 0.f;
        #pragma unroll
        for (int tt = 0; tt < 32; ++tt)
            d += __shfl(qown, tt, 32) * kp[tt];
        sc[j] = d * scale;
    }
    float mx = -1e30f;
    for (int j = 0; j < nch; ++j) mx = fmaxf(mx, sc[j]);
    #pragma unroll
    for (int off = 16; off >= 1; off >>= 1) mx = fmaxf(mx, __shfl_xor(mx, off, 32));
    float ssum = 0.f;
    for (int j = 0; j < nch; ++j) { sc[j] = __expf(sc[j] - mx); ssum += sc[j]; }
    #pragma unroll
    for (int off = 16; off >= 1; off >>= 1) ssum += __shfl_xor(ssum, off, 32);
    float inv = 1.f / ssum;
    float out = 0.f;
    for (int j = 0; j < nch; ++j) {
        const float* vrow = vb + (size_t)(j * 32) * DM + lane;
        #pragma unroll
        for (int tt = 0; tt < 32; ++tt) {
            float p = __shfl(sc[j], tt, 32);
            out += p * vrow[(size_t)tt * DM];
        }
    }
    O[(size_t)row * DM + hd * DH + lane] = out * inv;
}

// ---------------------------------------------------------------------------
// LayerNorm over 256: one wave per row. Out = act(LN(X [+ Res]) * g + b)
// Out may alias X. Out written at Out[row*ldo + c].
// ---------------------------------------------------------------------------
__global__ __launch_bounds__(128) void ln_kernel(
    const float* __restrict__ X, const float* __restrict__ Res,
    const float* __restrict__ g, const float* __restrict__ bta,
    float* __restrict__ Out, int rows, int ldo, int act)
{
    int task = blockIdx.x * 4 + (threadIdx.x >> 5);
    if (task >= rows) return;
    int lane = threadIdx.x & 31;
    const float* xp = X + (size_t)task * 256;
    const float* rp = Res ? Res + (size_t)task * 256 : nullptr;
    float v[8];
    float s = 0.f;
    #pragma unroll
    for (int i = 0; i < 8; ++i) {
        float x = xp[i * 32 + lane];
        if (rp) x += rp[i * 32 + lane];
        v[i] = x; s += x;
    }
    #pragma unroll
    for (int off = 16; off >= 1; off >>= 1) s += __shfl_xor(s, off, 32);
    float mean = s * (1.f / 256.f);
    float var = 0.f;
    #pragma unroll
    for (int i = 0; i < 8; ++i) { float d = v[i] - mean; var += d * d; }
    #pragma unroll
    for (int off = 16; off >= 1; off >>= 1) var += __shfl_xor(var, off, 32);
    var *= (1.f / 256.f);
    float rstd = rsqrtf(var + 1e-5f);
    #pragma unroll
    for (int i = 0; i < 8; ++i) {
        int c = i * 32 + lane;
        float o = (v[i] - mean) * rstd * g[c] + bta[c];
        if (act) o = fmaxf(o, 0.f);
        Out[(size_t)task * ldo + c] = o;
    }
}

// ---- elementwise helpers -------------------------------------------------

__global__ void add_kernel(const float* __restrict__ A, const float* __restrict__ Bv,
                           float* __restrict__ O, int n4)
{
    int idx = blockIdx.x * blockDim.x + threadIdx.x;
    if (idx >= n4) return;
    float4 a = ((const float4*)A)[idx];
    float4 b = ((const float4*)Bv)[idx];
    ((float4*)O)[idx] = make_float4(a.x + b.x, a.y + b.y, a.z + b.z, a.w + b.w);
}

// h/o gather: Out[b,p,:] = X[b, xk/yk, :]
__global__ void gather_tok(const float* __restrict__ X, float* __restrict__ Out, int useY)
{
    int idx = blockIdx.x * blockDim.x + threadIdx.x;   // over FROWS*64 float4s
    if (idx >= FROWS * 64) return;
    int row = idx >> 6, q = idx & 63;
    int b = row / NPAIR, p = row - b * NPAIR;
    int xx, yy; pair_xy(p, xx, yy);
    int tok = useY ? yy : xx;
    ((float4*)(Out + (size_t)row * 256))[q] =
        ((const float4*)(X + (size_t)(b * NTOK + tok) * 256))[q];
}

// Out = Feat + gather(bpe, xk/yk) + line_feat[b]
__global__ void add_pe_kernel(const float* __restrict__ Feat, const float* __restrict__ bpe,
                              const float* __restrict__ lf, float* __restrict__ Out, int useY)
{
    int idx = blockIdx.x * blockDim.x + threadIdx.x;
    if (idx >= FROWS * 64) return;
    int row = idx >> 6, q = idx & 63;
    int b = row / NPAIR, p = row - b * NPAIR;
    int xx, yy; pair_xy(p, xx, yy);
    int tok = useY ? yy : xx;
    float4 a = ((const float4*)(Feat + (size_t)row * 256))[q];
    float4 c = ((const float4*)(bpe + (size_t)(b * NTOK + tok) * 256))[q];
    float4 l = ((const float4*)(lf + (size_t)b * 256))[q];
    ((float4*)(Out + (size_t)row * 256))[q] =
        make_float4(a.x + c.x + l.x, a.y + c.y + l.y, a.z + c.z + l.z, a.w + c.w + l.w);
}

// Out[row, 0:256]=A[row], Out[row, 256:512]=B[row]
__global__ void concat2_kernel(const float* __restrict__ A, const float* __restrict__ Bp,
                               float* __restrict__ Out)
{
    int idx = blockIdx.x * blockDim.x + threadIdx.x;   // over FROWS*128 float4s
    if (idx >= FROWS * 128) return;
    int row = idx >> 7, q = idx & 127;
    float4 v = (q < 64) ? ((const float4*)(A + (size_t)row * 256))[q]
                        : ((const float4*)(Bp + (size_t)row * 256))[q - 64];
    ((float4*)(Out + (size_t)row * 512))[q] = v;
}

// mean over trailing 8x8 of line: one wave per (b, channel)
__global__ __launch_bounds__(128) void line_mean_kernel(const float* __restrict__ line,
                                                        float* __restrict__ out)
{
    int task = blockIdx.x * 4 + (threadIdx.x >> 5);
    if (task >= BATCH * 2048) return;
    int lane = threadIdx.x & 31;
    const float* p = line + (size_t)task * 64;
    float s = p[lane] + p[lane + 32];
    #pragma unroll
    for (int off = 16; off >= 1; off >>= 1) s += __shfl_xor(s, off, 32);
    if (lane == 0) out[task] = s * (1.f / 64.f);
}

// 36 spatial features for kept pairs (written at stride 64, zero padded)
__global__ void spatial36_kernel(const float* __restrict__ boxes,
                                 const float* __restrict__ sizes,
                                 float* __restrict__ out)
{
    int idx = blockIdx.x * blockDim.x + threadIdx.x;
    if (idx >= FROWS) return;
    int b = idx / NPAIR, p = idx - b * NPAIR;
    int x, y; pair_xy(p, x, y);
    const float* b1 = boxes + (size_t)(b * NTOK + x) * 4;
    const float* b2 = boxes + (size_t)(b * NTOK + y) * 4;
    float hh = sizes[b * 2 + 0], ww = sizes[b * 2 + 1];
    const float EPS = 1e-6f;
    float c1x = 0.5f * (b1[0] + b1[2]), c1y = 0.5f * (b1[1] + b1[3]);
    float c2x = 0.5f * (b2[0] + b2[2]), c2y = 0.5f * (b2[1] + b2[3]);
    float w1 = b1[2] - b1[0], h1 = b1[3] - b1[1];
    float w2 = b2[2] - b2[0], h2 = b2[3] - b2[1];
    float a1 = w1 * h1, a2 = w2 * h2;
    float dx = fabsf(c2x - c1x) / (w1 + EPS);
    float dy = fabsf(c2y - c1y) / (h1 + EPS);
    float ix = fmaxf(fminf(b1[2], b2[2]) - fmaxf(b1[0], b2[0]), 0.f);
    float iy = fmaxf(fminf(b1[3], b2[3]) - fmaxf(b1[1], b2[1]), 0.f);
    float inter = ix * iy;
    float iou = inter / (a1 + a2 - inter + EPS);
    float f[18];
    f[0] = c1x / ww;  f[1] = c1y / hh;  f[2] = c2x / ww;  f[3] = c2y / hh;
    f[4] = w1 / ww;   f[5] = h1 / hh;   f[6] = w2 / ww;   f[7] = h2 / hh;
    f[8] = a1 / (hh * ww); f[9] = a2 / (hh * ww);
    f[10] = a2 / (a1 + EPS); f[11] = w1 / (h1 + EPS); f[12] = w2 / (h2 + EPS);
    f[13] = iou;
    f[14] = (c2x > c1x) ? dx : 0.f;  f[15] = (c2x < c1x) ? dx : 0.f;
    f[16] = (c2y > c1y) ? dy : 0.f;  f[17] = (c2y < c1y) ? dy : 0.f;
    float* o = out + (size_t)idx * 64;
    #pragma unroll
    for (int j = 0; j < 18; ++j) { o[j] = f[j]; o[18 + j] = logf(f[j] + EPS); }
    #pragma unroll
    for (int j = 36; j < 64; ++j) o[j] = 0.f;
}

// zero-pad weight rows: dst (64 x n) = src (krows x n) padded with zeros
__global__ void pad_w_kernel(const float* __restrict__ src, float* __restrict__ dst,
                             int krows, int n)
{
    int idx = blockIdx.x * blockDim.x + threadIdx.x;
    if (idx >= 64 * n) return;
    int k = idx / n;
    dst[idx] = (k < krows) ? src[idx] : 0.f;
}

// sinusoidal box PE: 512 dims per (b, n) = [PE(cy) PE(cx) PE(h) PE(w)]
__global__ void boxpe_kernel(const float* __restrict__ boxes,
                             const float* __restrict__ sizes,
                             float* __restrict__ pe)
{
    int idx = blockIdx.x * blockDim.x + threadIdx.x;   // over 2048*512
    if (idx >= BATCH * NTOK * 512) return;
    int bn = idx >> 9, i = idx & 511;
    int b = bn >> 7;
    const float* bx = boxes + (size_t)bn * 4;
    float hh = sizes[b * 2 + 0], ww = sizes[b * 2 + 1];
    float x0 = bx[0] / ww, y0 = bx[1] / hh, x1 = bx[2] / ww, y1 = bx[3] / hh;
    int grp = i >> 7, j = i & 127, k = j >> 1;
    float p = (grp == 0) ? 0.5f * (y0 + y1)
            : (grp == 1) ? 0.5f * (x0 + x1)
            : (grp == 2) ? (y1 - y0)
                         : (x1 - x0);
    float t = __powf(20.f, (float)k * (1.f / 64.f));
    float arg = p * 6.283185307179586f / t;
    pe[idx] = (j & 1) ? __cosf(arg) : __sinf(arg);
}

// ---------------------------------------------------------------------------

static inline void launch_gemm(hipStream_t s, const float* A, int lda, const float* W,
                               const float* bias, float* C, int rows, int K,
                               int N, int ldw, int ldc, int act)
{
    dim3 g(N / 64, (rows + 63) / 64);
    gemm_wmma<<<g, 128, 0, s>>>(A, W, bias, C, rows, K, N, lda, ldw, ldc, act);
}
static inline int cdiv(int a, int b) { return (a + b - 1) / b; }

extern "C" void kernel_launch(void* const* d_in, const int* in_sizes, int n_in,
                              void* d_out, int out_size, void* d_ws, size_t ws_size,
                              hipStream_t stream)
{
    (void)in_sizes; (void)n_in; (void)out_size; (void)ws_size;
    const float* boxes  = (const float*)d_in[0];
    const float* embeds = (const float*)d_in[1];
    const float* sizes  = (const float*)d_in[2];
    const float* pose   = (const float*)d_in[3];
    const float* line   = (const float*)d_in[4];
    const float *sw1=(const float*)d_in[5],  *sb1=(const float*)d_in[6];
    const float *sw2=(const float*)d_in[7],  *sb2=(const float*)d_in[8];
    const float *sw3=(const float*)d_in[9],  *sb3=(const float*)d_in[10];
    const float *bw =(const float*)d_in[11], *bb =(const float*)d_in[12];
    const float *pw =(const float*)d_in[13], *pb =(const float*)d_in[14];
    const float *lw =(const float*)d_in[15], *lb =(const float*)d_in[16];
    const float *enc_pe_w=(const float*)d_in[17], *enc_pe_b=(const float*)d_in[18];
    const float *enc_qkv_w=(const float*)d_in[19], *enc_qkv_b=(const float*)d_in[20];
    const float *enc_out_w=(const float*)d_in[21], *enc_out_b=(const float*)d_in[22];
    const float *enc_ff1_w=(const float*)d_in[23], *enc_ff1_b=(const float*)d_in[24];
    const float *enc_ff2_w=(const float*)d_in[25], *enc_ff2_b=(const float*)d_in[26];
    const float *enc_ln1_g=(const float*)d_in[27], *enc_ln1_b=(const float*)d_in[28];
    const float *enc_ln2_g=(const float*)d_in[29], *enc_ln2_b=(const float*)d_in[30];
    const float *dec_q_w=(const float*)d_in[31],  *dec_q_b=(const float*)d_in[32];
    const float *dec_kv_w=(const float*)d_in[33], *dec_kv_b=(const float*)d_in[34];
    const float *dec_out_w=(const float*)d_in[35],*dec_out_b=(const float*)d_in[36];
    const float *dec_ff1_w=(const float*)d_in[37],*dec_ff1_b=(const float*)d_in[38];
    const float *dec_ff2_w=(const float*)d_in[39],*dec_ff2_b=(const float*)d_in[40];
    const float *dec_ln1_g=(const float*)d_in[41],*dec_ln1_b=(const float*)d_in[42];
    const float *dec_ln2_g=(const float*)d_in[43],*dec_ln2_b=(const float*)d_in[44];
    const float *fc1_w=(const float*)d_in[45], *fc1_b=(const float*)d_in[46];
    const float *fc2_w=(const float*)d_in[47], *fc2_b=(const float*)d_in[48];
    const float *mln1_g=(const float*)d_in[49], *mln1_b=(const float*)d_in[50];
    const float *mln2_g=(const float*)d_in[51], *mln2_b=(const float*)d_in[52];
    const float *m1_w=(const float*)d_in[53], *m1_b=(const float*)d_in[54];
    const float *m2_w=(const float*)d_in[55], *m2_b=(const float*)d_in[56];

    float* ws = (float*)d_ws;
    size_t off = 0;
    const size_t F = FROWS;
    float* zcat = ws + off; off += F * 512;
    float* hbuf = ws + off; off += F * 256;
    float* obuf = ws + off; off += F * 256;
    float* Dbuf = ws + off; off += F * 256;
    float* Sar  = ws + off; off += F * 1024;      // big arena
    float* box_pe = ws + off; off += 2048 * 512;
    float* posx   = ws + off; off += 2048 * 256;
    float* bpe    = ws + off; off += 2048 * 256;
    float* xbuf   = ws + off; off += 2048 * 256;
    float* xpbuf  = ws + off; off += 2048 * 256;
    float* qe     = ws + off; off += 2048 * 256;
    float* ke     = ws + off; off += 2048 * 256;
    float* ve     = ws + off; off += 2048 * 256;
    float* aoe    = ws + off; off += 2048 * 256;
    float* tmpe   = ws + off; off += 2048 * 256;
    float* ffe    = ws + off; off += 2048 * 1024;
    float* pctx   = ws + off; off += 1024 * 256;
    float* kbuf   = ws + off; off += 1024 * 256;
    float* vbuf   = ws + off; off += 1024 * 256;
    float* lmean  = ws + off; off += 16 * 2048;
    float* lfeat  = ws + off; off += 16 * 256;
    float* sw1p   = ws + off; off += 64 * 128;    // zero-padded sw1 (64x128)

    float* Q0 = Sar;
    float* Q1 = Sar + F * 256;
    float* Q2 = Sar + F * 512;
    float* Q3 = Sar + F * 768;
    const float scale = 0.17677669529663687f;   // 1/sqrt(32)

    // ---- spatial path: f36(pad64) -> MLP -> fc2 -> LN+relu into zcat[:,256:512]
    pad_w_kernel<<<cdiv(64 * 128, 256), 256, 0, stream>>>(sw1, sw1p, 36, 128);
    spatial36_kernel<<<cdiv(FROWS, 256), 256, 0, stream>>>(boxes, sizes, Q0);
    launch_gemm(stream, Q0, 64, sw1p, sb1, Q1, FROWS, 64, 128, 128, 128, 1);
    launch_gemm(stream, Q1, 128, sw2, sb2, Q2, FROWS, 128, 256, 256, 256, 1);
    launch_gemm(stream, Q2, 256, sw3, sb3, Q3, FROWS, 256, 256, 256, 256, 1);
    launch_gemm(stream, Q3, 256, fc2_w, fc2_b, Q0, FROWS, 256, 256, 256, 256, 0);
    ln_kernel<<<cdiv(FROWS, 4), 128, 0, stream>>>(Q0, nullptr, mln2_g, mln2_b,
                                                  zcat + 256, FROWS, 512, 1);

    // ---- line features
    line_mean_kernel<<<cdiv(BATCH * 2048, 4), 128, 0, stream>>>(line, lmean);
    launch_gemm(stream, lmean, 2048, lw, lb, lfeat, BATCH, 2048, 256, 256, 256, 0);

    // ---- box PE, projections
    boxpe_kernel<<<cdiv(2048 * 512, 256), 256, 0, stream>>>(boxes, sizes, box_pe);
    launch_gemm(stream, box_pe, 512, enc_pe_w, enc_pe_b, posx, 2048, 512, 256, 256, 256, 0);
    launch_gemm(stream, box_pe, 512, bw, bb, bpe, 2048, 512, 256, 256, 256, 0);

    // ---- encoder
    hipMemcpyAsync(xbuf, embeds, (size_t)2048 * 256 * sizeof(float),
                   hipMemcpyDeviceToDevice, stream);
    for (int i = 0; i < 2; ++i) {
        add_kernel<<<cdiv(2048 * 64, 256), 256, 0, stream>>>(xbuf, posx, xpbuf, 2048 * 64);
        const float* Wqkv = enc_qkv_w + (size_t)i * 256 * 768;
        const float* Bqkv = enc_qkv_b + (size_t)i * 768;
        launch_gemm(stream, xpbuf, 256, Wqkv,       Bqkv,       qe, 2048, 256, 256, 768, 256, 0);
        launch_gemm(stream, xpbuf, 256, Wqkv + 256, Bqkv + 256, ke, 2048, 256, 256, 768, 256, 0);
        launch_gemm(stream, xbuf,  256, Wqkv + 512, Bqkv + 512, ve, 2048, 256, 256, 768, 256, 0);
        attn_kernel<<<cdiv(2048 * NHEADS, 4), 128, 0, stream>>>(qe, ke, ve, aoe,
                                                                2048, 128, 128, scale);
        launch_gemm(stream, aoe, 256, enc_out_w + (size_t)i * 65536, enc_out_b + i * 256,
                    tmpe, 2048, 256, 256, 256, 256, 0);
        ln_kernel<<<cdiv(2048, 4), 128, 0, stream>>>(xbuf, tmpe, enc_ln1_g + i * 256,
                                                     enc_ln1_b + i * 256, xbuf, 2048, 256, 0);
        launch_gemm(stream, xbuf, 256, enc_ff1_w + (size_t)i * 262144, enc_ff1_b + i * 1024,
                    ffe, 2048, 256, 1024, 1024, 1024, 1);
        launch_gemm(stream, ffe, 1024, enc_ff2_w + (size_t)i * 262144, enc_ff2_b + i * 256,
                    tmpe, 2048, 1024, 256, 256, 256, 0);
        ln_kernel<<<cdiv(2048, 4), 128, 0, stream>>>(xbuf, tmpe, enc_ln2_g + i * 256,
                                                     enc_ln2_b + i * 256, xbuf, 2048, 256, 0);
    }

    // ---- pose context, gathers
    launch_gemm(stream, pose, 768, pw, pb, pctx, BATCH * MPOSE, 768, 256, 256, 256, 0);
    gather_tok<<<cdiv(FROWS * 64, 256), 256, 0, stream>>>(xbuf, hbuf, 0);
    gather_tok<<<cdiv(FROWS * 64, 256), 256, 0, stream>>>(xbuf, obuf, 1);

    // ---- decoder (2 layers, h and o streams)
    for (int l = 0; l < 2; ++l) {
        const float* Wkv = dec_kv_w + (size_t)l * 131072;
        const float* Bkv = dec_kv_b + (size_t)l * 512;
        launch_gemm(stream, pctx, 256, Wkv,       Bkv,       kbuf, 1024, 256, 256, 512, 256, 0);
        launch_gemm(stream, pctx, 256, Wkv + 256, Bkv + 256, vbuf, 1024, 256, 256, 512, 256, 0);
        for (int st = 0; st < 2; ++st) {
            float* feat = st ? obuf : hbuf;
            add_pe_kernel<<<cdiv(FROWS * 64, 256), 256, 0, stream>>>(feat, bpe, lfeat,
                                                                     Dbuf, st);
            launch_gemm(stream, Dbuf, 256, dec_q_w + (size_t)l * 65536, dec_q_b + l * 256,
                        Q0, FROWS, 256, 256, 256, 256, 0);
            attn_kernel<<<cdiv(FROWS * NHEADS, 4), 128, 0, stream>>>(Q0, kbuf, vbuf, Q1,
                                                                     FROWS, NPAIR, 64, scale);
            launch_gemm(stream, Q1, 256, dec_out_w + (size_t)l * 65536, dec_out_b + l * 256,
                        Q2, FROWS, 256, 256, 256, 256, 0);
            ln_kernel<<<cdiv(FROWS, 4), 128, 0, stream>>>(feat, Q2, dec_ln1_g + l * 256,
                                                          dec_ln1_b + l * 256, feat,
                                                          FROWS, 256, 0);
            launch_gemm(stream, feat, 256, dec_ff1_w + (size_t)l * 262144, dec_ff1_b + l * 1024,
                        Sar, FROWS, 256, 1024, 1024, 1024, 1);
            launch_gemm(stream, Sar, 1024, dec_ff2_w + (size_t)l * 262144, dec_ff2_b + l * 256,
                        Dbuf, FROWS, 1024, 256, 256, 256, 0);
            ln_kernel<<<cdiv(FROWS, 4), 128, 0, stream>>>(feat, Dbuf, dec_ln2_g + l * 256,
                                                          dec_ln2_b + l * 256, feat,
                                                          FROWS, 256, 0);
        }
    }

    // ---- head: ho=LN(cat(h,o)@fc1) relu -> zcat[:, :256]; m1; m2 -> out
    concat2_kernel<<<cdiv(FROWS * 128, 256), 256, 0, stream>>>(hbuf, obuf, Sar);
    launch_gemm(stream, Sar, 512, fc1_w, fc1_b, Dbuf, FROWS, 512, 256, 256, 256, 0);
    ln_kernel<<<cdiv(FROWS, 4), 128, 0, stream>>>(Dbuf, nullptr, mln1_g, mln1_b,
                                                  zcat, FROWS, 512, 1);
    launch_gemm(stream, zcat, 512, m1_w, m1_b, Sar, FROWS, 512, 384, 384, 384, 1);
    launch_gemm(stream, Sar, 384, m2_w, m2_b, (float*)d_out, FROWS, 384, 256, 256, 256, 1);
}
